// RWKV_TimeMix_24163486007378
// MI455X (gfx1250) — compile-verified, matrix operand order checked by reading, not run on hardware
//
#include <hip/hip_runtime.h>
#include <hip/hip_bf16.h>

// ---------------------------------------------------------------------------
// RWKV-style TimeMix for MI455X (gfx1250, wave32, WMMA).
//   r = sigmoid(x Wr^T); k = x Wk^T; v = x Wv^T      (bf16 WMMA GEMMs, dbl-buffered LDS)
//   s_t = d*s_{t-1} + k_t*v_t ; o = r*s              (3-pass chunked parallel scan)
//   GroupNorm(1 group over (T,D) per batch) + affine (fused stats + apply)
//   out = normed Wo^T                                (bf16 WMMA GEMM)
// GEMM geometry: block tile 128x256, BK=32, 8 waves (2x4), warp tile 64x64
//   -> 16 v_wmma_f32_16x16x32_bf16 per wave per K-step, 16 ds_load_b128.
// ---------------------------------------------------------------------------

typedef __attribute__((ext_vector_type(16))) __bf16 v16bf;
typedef __attribute__((ext_vector_type(8)))  __bf16 v8bf;
typedef __attribute__((ext_vector_type(4)))  __bf16 v4bf;
typedef __attribute__((ext_vector_type(8)))  float  v8f;

#define BM 128
#define BN 256
#define BK 32
#define LDSS (BK + 8)   // padded LDS row stride in bf16 (80 B): conflict-free b128 frag reads

__device__ __forceinline__ float fast_sigmoid(float x) {
  return __builtin_amdgcn_rcpf(1.0f + __expf(-x));   // v_exp_f32 + v_rcp_f32
}

// One 16x32 bf16 WMMA operand fragment from a padded row-major LDS tile.
// Lane l (0-15): row l, K {0..7,16..23}; lane l+16: row l, K {8..15,24..31}.
__device__ __forceinline__ v16bf load_frag(const __bf16* base, int row, int half) {
  union { v16bf v; v8bf h[2]; } u;
  const __bf16* p = base + row * LDSS + half * 8;
  u.h[0] = *(const v8bf*)(p);
  u.h[1] = *(const v8bf*)(p + 16);
  return u.v;
}

// C[M,N] (f32) = A[M,K] bf16 row-major  x  Bw[N,K]^T bf16 row-major. epilogue 1 = sigmoid.
__global__ __launch_bounds__(256) void gemm_bf16_nt(
    const __bf16* __restrict__ A,
    const __bf16* __restrict__ Bw,
    float* __restrict__ C,
    int M, int N, int K, int epilogue)
{
  __shared__ __bf16 lA[2][BM * LDSS];
  __shared__ __bf16 lB[2][BN * LDSS];

  const int tid   = threadIdx.x;
  const int lane  = tid & 31;
  const int wave  = tid >> 5;
  const int half  = lane >> 4;
  const int l     = lane & 15;
  const int waveM = wave >> 2;       // 0..1 : 64 rows
  const int waveN = wave & 3;        // 0..3 : 64 cols

  const int bM = blockIdx.x * BM;
  const int bN = blockIdx.y * BN;

  // Cooperative tile loads (16B chunks): A 128x32 = 512 chunks (2/thread),
  // B 256x32 = 1024 chunks (4/thread).
  const int cr = tid >> 2;            // base row 0..63
  const int cc = (tid & 3) * 8;       // K-chunk 0/8/16/24

  const __bf16* gA0 = A  + (size_t)(bM + cr)       * K + cc;
  const __bf16* gA1 = A  + (size_t)(bM + cr + 64)  * K + cc;
  const __bf16* gB0 = Bw + (size_t)(bN + cr)       * K + cc;
  const __bf16* gB1 = Bw + (size_t)(bN + cr + 64)  * K + cc;
  const __bf16* gB2 = Bw + (size_t)(bN + cr + 128) * K + cc;
  const __bf16* gB3 = Bw + (size_t)(bN + cr + 192) * K + cc;

  v8f acc[4][4];
  #pragma unroll
  for (int i = 0; i < 4; i++)
    #pragma unroll
    for (int j = 0; j < 4; j++)
      acc[i][j] = v8f{};

  const int nsteps = K / BK;

  // Preload stage 0
  {
    uint4 a0 = *(const uint4*)(gA0);
    uint4 a1 = *(const uint4*)(gA1);
    uint4 b0 = *(const uint4*)(gB0);
    uint4 b1 = *(const uint4*)(gB1);
    uint4 b2 = *(const uint4*)(gB2);
    uint4 b3 = *(const uint4*)(gB3);
    *(uint4*)(&lA[0][(cr)       * LDSS + cc]) = a0;
    *(uint4*)(&lA[0][(cr + 64)  * LDSS + cc]) = a1;
    *(uint4*)(&lB[0][(cr)       * LDSS + cc]) = b0;
    *(uint4*)(&lB[0][(cr + 64)  * LDSS + cc]) = b1;
    *(uint4*)(&lB[0][(cr + 128) * LDSS + cc]) = b2;
    *(uint4*)(&lB[0][(cr + 192) * LDSS + cc]) = b3;
  }

  for (int s = 0; s < nsteps; s++) {
    __syncthreads();   // stage s&1 visible; stage (s+1)&1 free to overwrite

    const bool more = (s + 1) < nsteps;
    uint4 a0, a1, b0, b1, b2, b3;
    if (more) {        // issue next-tile global loads before compute (overlap)
      const int kn = (s + 1) * BK;
      a0 = *(const uint4*)(gA0 + kn);
      a1 = *(const uint4*)(gA1 + kn);
      b0 = *(const uint4*)(gB0 + kn);
      b1 = *(const uint4*)(gB1 + kn);
      b2 = *(const uint4*)(gB2 + kn);
      b3 = *(const uint4*)(gB3 + kn);
      if (s + 2 < nsteps) {   // pull the tile after that toward L2
        __builtin_prefetch(gA0 + kn + BK, 0, 1);
        __builtin_prefetch(gB0 + kn + BK, 0, 1);
        __builtin_prefetch(gB2 + kn + BK, 0, 1);
      }
    }

    const __bf16* cA = &lA[s & 1][0];
    const __bf16* cB = &lB[s & 1][0];

    v16bf af[4];
    #pragma unroll
    for (int i = 0; i < 4; i++) af[i] = load_frag(cA, waveM * 64 + i * 16 + l, half);

    #pragma unroll
    for (int j = 0; j < 4; j++) {
      const v16bf bfr = load_frag(cB, waveN * 64 + j * 16 + l, half);
      #pragma unroll
      for (int i = 0; i < 4; i++)
        acc[i][j] = __builtin_amdgcn_wmma_f32_16x16x32_bf16(
            false, af[i], false, bfr, (short)0, acc[i][j], false, false);
    }

    if (more) {
      __bf16* nA = &lA[(s + 1) & 1][0];
      __bf16* nB = &lB[(s + 1) & 1][0];
      *(uint4*)(&nA[(cr)       * LDSS + cc]) = a0;
      *(uint4*)(&nA[(cr + 64)  * LDSS + cc]) = a1;
      *(uint4*)(&nB[(cr)       * LDSS + cc]) = b0;
      *(uint4*)(&nB[(cr + 64)  * LDSS + cc]) = b1;
      *(uint4*)(&nB[(cr + 128) * LDSS + cc]) = b2;
      *(uint4*)(&nB[(cr + 192) * LDSS + cc]) = b3;
    }
  }

  // C/D layout: VGPR v, lanes 0-15 -> (M=v, N=lane); lanes 16-31 -> (M=v+8, N=lane-16)
  #pragma unroll
  for (int i = 0; i < 4; i++) {
    #pragma unroll
    for (int j = 0; j < 4; j++) {
      const int col   = bN + waveN * 64 + j * 16 + l;
      const int rbase = bM + waveM * 64 + i * 16 + half * 8;
      #pragma unroll
      for (int v = 0; v < 8; v++) {
        float o = acc[i][j][v];
        if (epilogue == 1) o = fast_sigmoid(o);
        C[(size_t)(rbase + v) * N + col] = o;
      }
    }
  }
}

// f32 -> bf16, 4 elements/thread
__global__ void cvt_f32_bf16(const float* __restrict__ in, __bf16* __restrict__ out, int n4) {
  int i = blockIdx.x * blockDim.x + threadIdx.x;
  if (i < n4) {
    float4 f = ((const float4*)in)[i];
    v4bf v = { (__bf16)f.x, (__bf16)f.y, (__bf16)f.z, (__bf16)f.w };
    *(v4bf*)(out + (size_t)i * 4) = v;
  }
}

__global__ void zero_f32(float* p, int n) {
  int i = blockIdx.x * blockDim.x + threadIdx.x;
  if (i < n) p[i] = 0.0f;
}

// ---- 3-pass chunked decay scan -------------------------------------------
// Pass A: per (b, e, chunk) local scan. Sloc may alias Kp (same-index RAW per thread).
__global__ void scan_chunks(const float* Kp, const float* Vp,
                            const float* __restrict__ decay,
                            float* Sloc, float* __restrict__ totals,
                            int T, int D, int L)
{
  const int e  = blockIdx.x * blockDim.x + threadIdx.x;
  const int c  = blockIdx.y;
  const int b  = blockIdx.z;
  const int nc = gridDim.y;
  if (e >= D) return;
  const float d = fast_sigmoid(decay[e]);
  const size_t base = ((size_t)b * T + (size_t)c * L) * D + e;
  float s = 0.0f;
  for (int j = 0; j < L; j++) {
    const size_t idx = base + (size_t)j * D;
    const float kv = Kp[idx] * Vp[idx];
    s = fmaf(d, s, kv);
    Sloc[idx] = s;
  }
  totals[((size_t)b * nc + c) * D + e] = s;
}

// Pass B: carry recurrence across chunks: carry_{c+1} = total_c + d^L * carry_c.
__global__ void scan_carry(const float* __restrict__ decay,
                           const float* __restrict__ totals,
                           float* __restrict__ carry,
                           int D, int nc, int L)
{
  const int e = blockIdx.x * blockDim.x + threadIdx.x;
  const int b = blockIdx.y;
  if (e >= D) return;
  const float d  = fast_sigmoid(decay[e]);
  const float dL = __powf(d, (float)L);
  float cy = 0.0f;
  for (int c = 0; c < nc; c++) {
    const size_t i = ((size_t)b * nc + c) * D + e;
    carry[i] = cy;
    cy = totals[i] + dL * cy;
  }
}

// Pass C: s = s_loc + d^{j+1} * carry; o = r*s; fused GroupNorm stats.
// S may alias R (same-index RAW per thread).
__global__ __launch_bounds__(256) void scan_finalize(
    const float* R, const float* __restrict__ Sloc,
    const float* __restrict__ carry, const float* __restrict__ decay,
    float* S, float* __restrict__ stats,
    int D, int Lbits, int nc)
{
  __shared__ float redS[8], redQ[8];
  const int b = blockIdx.y;
  const int i = blockIdx.x * blockDim.x + threadIdx.x;   // index within T*D
  const int e = i & (D - 1);
  const int t = i / D;
  const int c = t >> Lbits;
  const int j = t & ((1 << Lbits) - 1);
  const size_t idx = (size_t)b * gridDim.x * blockDim.x + i;

  const float d  = fast_sigmoid(decay[e]);
  const float cy = carry[((size_t)b * nc + c) * D + e];
  const float s  = Sloc[idx] + __powf(d, (float)(j + 1)) * cy;
  const float o  = R[idx] * s;
  S[idx] = o;

  // wave -> block reduction of (sum, sumsq), then 2 atomics per block
  float sum = o, sq = o * o;
  #pragma unroll
  for (int off = 16; off > 0; off >>= 1) {
    sum += __shfl_down(sum, off, 32);
    sq  += __shfl_down(sq,  off, 32);
  }
  const int lane = threadIdx.x & 31, wv = threadIdx.x >> 5;
  if (lane == 0) { redS[wv] = sum; redQ[wv] = sq; }
  __syncthreads();
  if (threadIdx.x == 0) {
    float ts = 0.0f, tq = 0.0f;
    #pragma unroll
    for (int w = 0; w < 8; w++) { ts += redS[w]; tq += redQ[w]; }
    atomicAdd(&stats[2 * b + 0], ts);
    atomicAdd(&stats[2 * b + 1], tq);
  }
}

// GroupNorm apply + per-channel affine -> bf16 for the final GEMM.
__global__ void gn_apply(const float* __restrict__ S, const float* __restrict__ stats,
                         const float* __restrict__ gw, const float* __restrict__ gb,
                         __bf16* __restrict__ out, int TD, int Dmask)
{
  const int b = blockIdx.y;
  const int i = blockIdx.x * blockDim.x + threadIdx.x;
  if (i >= TD) return;
  const float invN = 1.0f / (float)TD;
  const float mean = stats[2 * b + 0] * invN;
  const float var  = stats[2 * b + 1] * invN - mean * mean;
  const float rstd = rsqrtf(var + 1e-5f);
  const int e = i & Dmask;
  const size_t idx = (size_t)b * TD + i;
  const float o = (S[idx] - mean) * rstd * gw[e] + gb[e];
  out[idx] = (__bf16)o;
}

extern "C" void kernel_launch(void* const* d_in, const int* in_sizes, int n_in,
                              void* d_out, int out_size, void* d_ws, size_t ws_size,
                              hipStream_t stream) {
  constexpr int T = 2048, D = 2048;
  constexpr int Lbits = 7, L = 1 << Lbits, nc = T / L;   // 16 chunks of 128
  const int B = in_sizes[0] / (T * D);
  const int M = B * T, N = D, K = D;

  const float* x     = (const float*)d_in[0];
  const float* Wr    = (const float*)d_in[1];
  const float* Wk    = (const float*)d_in[2];
  const float* Wv    = (const float*)d_in[3];
  const float* Wo    = (const float*)d_in[4];
  const float* decay = (const float*)d_in[5];
  const float* gnw   = (const float*)d_in[6];
  const float* gnb   = (const float*)d_in[7];
  float* outp = (float*)d_out;

  // Workspace carve-up (~257 MB with aliasing)
  unsigned char* ws = (unsigned char*)d_ws;
  __bf16* xb  = (__bf16*)ws;  ws += (size_t)M * K * sizeof(__bf16);  // reused for normed bf16
  __bf16* wrb = (__bf16*)ws;  ws += (size_t)N * K * sizeof(__bf16);
  __bf16* wkb = (__bf16*)ws;  ws += (size_t)N * K * sizeof(__bf16);
  __bf16* wvb = (__bf16*)ws;  ws += (size_t)N * K * sizeof(__bf16);
  __bf16* wob = (__bf16*)ws;  ws += (size_t)N * K * sizeof(__bf16);
  float*  Rb  = (float*)ws;   ws += (size_t)M * N * sizeof(float);   // reused as scan output S
  float*  Kb  = (float*)ws;   ws += (size_t)M * N * sizeof(float);   // reused in-place as Sloc
  float*  Vb  = (float*)ws;   ws += (size_t)M * N * sizeof(float);
  float*  totals = (float*)ws; ws += (size_t)B * nc * D * sizeof(float);
  float*  carry  = (float*)ws; ws += (size_t)B * nc * D * sizeof(float);
  float*  stats  = (float*)ws;                                       // 2*B floats

  // 1) f32 -> bf16 conversions
  {
    int n4 = (M * K) / 4;
    cvt_f32_bf16<<<(n4 + 255) / 256, 256, 0, stream>>>(x, xb, n4);
    n4 = (N * K) / 4;
    cvt_f32_bf16<<<(n4 + 255) / 256, 256, 0, stream>>>(Wr, wrb, n4);
    cvt_f32_bf16<<<(n4 + 255) / 256, 256, 0, stream>>>(Wk, wkb, n4);
    cvt_f32_bf16<<<(n4 + 255) / 256, 256, 0, stream>>>(Wv, wvb, n4);
    cvt_f32_bf16<<<(n4 + 255) / 256, 256, 0, stream>>>(Wo, wob, n4);
  }
  zero_f32<<<1, 32, 0, stream>>>(stats, 2 * B);

  // 2) projections (WMMA bf16 GEMMs)
  dim3 ggrid(M / BM, N / BN);
  gemm_bf16_nt<<<ggrid, 256, 0, stream>>>(xb, wrb, Rb, M, N, K, /*sigmoid*/1);
  gemm_bf16_nt<<<ggrid, 256, 0, stream>>>(xb, wkb, Kb, M, N, K, 0);
  gemm_bf16_nt<<<ggrid, 256, 0, stream>>>(xb, wvb, Vb, M, N, K, 0);

  // 3) chunked decay scan (Sloc in-place over Kb; S in-place over Rb) + GN stats
  scan_chunks<<<dim3(D / 256, nc, B), 256, 0, stream>>>(Kb, Vb, decay, Kb, totals, T, D, L);
  scan_carry<<<dim3(D / 256, B), 256, 0, stream>>>(decay, totals, carry, D, nc, L);
  scan_finalize<<<dim3((T * D) / 256, B), 256, 0, stream>>>(Rb, Kb, carry, decay, Rb, stats,
                                                            D, Lbits, nc);

  // 4) GroupNorm apply -> bf16 (reuse xb)
  gn_apply<<<dim3((T * D) / 256, B), 256, 0, stream>>>(Rb, stats, gnw, gnb, xb, T * D, D - 1);

  // 5) output projection
  gemm_bf16_nt<<<ggrid, 256, 0, stream>>>(xb, wob, outp, M, N, K, 0);
}
